// MTGRU_42820823941604
// MI455X (gfx1250) — compile-verified, matrix operand order checked by reading, not run on hardware
//
#include <hip/hip_runtime.h>
#include <math.h>

// ---- problem constants ----
constexpr int kH    = 512;
constexpr int kH3   = 1536;
constexpr int kIN   = 64;
constexpr int kOUT  = 64;
constexpr int kK    = 16;    // latent
constexpr int kNH   = 300;   // NPSI_H
constexpr int kNLR  = 30;    // NPSI_LR
constexpr int kB    = 64;
constexpr int kT    = 128;
constexpr int kNPSI = 919104;
constexpr int OFF_BH  = 786432;           // start of non-Whh psi region
constexpr int kWhhElems = 786432;         // 512*1536
constexpr int kWhhTiles = 49152;          // 786432/16
// compact psi2 buffer (everything after Whh), offsets relative to OFF_BH
constexpr int P2_BH  = 0;                 // 1536
constexpr int P2_WIH = 1536;              // 64*1536
constexpr int P2_C   = 99840;             // 512*64
constexpr int P2_DV  = 132608;            // 64
constexpr int kPsi2  = 132672;
constexpr int kStagePitch = 520;          // 512 + 8 halves pad (bank-conflict relief)

typedef __bf16 bf16;
typedef __attribute__((ext_vector_type(16))) __bf16 v16bf;
typedef __attribute__((ext_vector_type(8)))  float  v8f;

__device__ __forceinline__ float sigf(float x) { return 1.0f / (1.0f + __expf(-x)); }

__device__ __forceinline__ v8f wmma_bf(v16bf a, v16bf b, v8f c) {
  return __builtin_amdgcn_wmma_f32_16x16x32_bf16(false, a, false, b, (short)0, c, false, false);
}

// A tile (16x32 bf16), rows contiguous (strideK==1).
// lanes 0-15 & 16-31 hold rows M=0..15; lane<16 -> K {0..7,16..23}, lane>=16 -> K {8..15,24..31}
__device__ __forceinline__ v16bf load_a16(const bf16* base, int m0, int k0, int strideM) {
  int lane = threadIdx.x & 31;
  int m = lane & 15;
  int ko = (lane < 16) ? 0 : 8;
  const bf16* p = base + (size_t)(m0 + m) * strideM + k0;
  v16bf r;
#pragma unroll
  for (int i = 0; i < 8; ++i) r[i] = p[ko + i];
#pragma unroll
  for (int i = 0; i < 8; ++i) r[8 + i] = p[ko + 16 + i];
  return r;
}

// Generic-stride B tile (32x16): B[k][n] = base[(n0+n)*sN + (k0+k)*sK]  (cold paths only)
__device__ __forceinline__ v16bf load_b16(const bf16* base, int k0, int n0, int sK, int sN) {
  int lane = threadIdx.x & 31;
  int n = lane & 15;
  int ko = (lane < 16) ? 0 : 8;
  const bf16* p = base + (size_t)(n0 + n) * sN;
  v16bf r;
#pragma unroll
  for (int i = 0; i < 8; ++i) r[i] = p[(size_t)(k0 + ko + i) * sK];
#pragma unroll
  for (int i = 0; i < 8; ++i) r[8 + i] = p[(size_t)(k0 + ko + 16 + i) * sK];
  return r;
}

// Swizzled B tile: each lane's 16 halves are one contiguous 32B chunk.
// layout: element = base[ ((nc*16 + kc)*32 + lane)*16 + h ]
__device__ __forceinline__ v16bf load_b_swz(const bf16* base, int kc, int nc) {
  int lane = threadIdx.x & 31;
  return *(const v16bf*)(base + ((size_t)((nc * 16 + kc) * 32 + lane) << 4));
}

// ---- kernel 1: latent -> r[64x30] padded to 32, bf16 ----
__global__ void hyper_kernel(const float* mu, const float* sd, const float* eps,
                             const float* W1, const float* b1,
                             const float* W2, const float* b2, bf16* r_bf) {
  __shared__ float zs[kK];
  __shared__ float hid[kNH];
  int b = blockIdx.x, tid = threadIdx.x;
  if (tid < kK) zs[tid] = mu[b * kK + tid] + eps[b * kK + tid] * sd[b * kK + tid];
  __syncthreads();
  for (int n = tid; n < kNH; n += blockDim.x) {
    float a = b1[n];
    for (int k = 0; k < kK; ++k) a += zs[k] * W1[k * kNH + n];
    hid[n] = tanhf(a);
  }
  __syncthreads();
  if (tid < 32) {
    float a = 0.0f;
    if (tid < kNLR) {
      a = b2[tid];
      for (int n = 0; n < kNH; ++n) a += hid[n] * W2[n * kNLR + tid];
    }
    r_bf[b * 32 + tid] = (bf16)a;
  }
}

// ---- fp32 -> bf16 convert ----
__global__ void cvt_kernel(const float* src, bf16* dst, int n) {
  for (int i = blockIdx.x * blockDim.x + threadIdx.x; i < n; i += gridDim.x * blockDim.x)
    dst[i] = (bf16)src[i];
}

// ---- repack Whh2 (fp32 [1536][512], gh[j]=sum_i h[i]*Whh2[j][i]) into swizzled bf16 ----
__global__ void repack2_kernel(const float* Whh2, bf16* dst) {
  for (int idx = blockIdx.x * blockDim.x + threadIdx.x; idx < kWhhElems;
       idx += gridDim.x * blockDim.x) {
    int h    = idx & 15;
    int lane = (idx >> 4) & 31;
    int t    = idx >> 9;
    int kc   = t & 15;
    int nc   = t >> 4;
    int nl   = lane & 15;
    int ko   = (lane < 16) ? 0 : 8;
    int kr   = ko + ((h < 8) ? h : (16 + h - 8));
    int k    = kc * 32 + kr;          // i (hidden index)
    int n    = nc * 16 + nl;          // j (gate index)
    dst[idx] = (bf16)Whh2[(size_t)n * kH + k];
  }
}

// ---- kernel 2a: psi Whh region -> swizzled layout, LDS-staged coalesced stores ----
// one block (256 thr, 8 waves) per swizzle super-tile (nc,kc): 32 k-rows x 16 cols x 64 samples
// grid = 96*16 = 1536
__global__ void __launch_bounds__(256) psi_whh_kernel(const bf16* r_bf, const float* W3,
                                                      const float* b3, bf16* whhswz) {
  __shared__ bf16 stage[kB * kStagePitch];      // ~65 KB
  int wave = threadIdx.x >> 5;
  int lane = threadIdx.x & 31;
  int jl = lane & 15;
  int kc = blockIdx.x & 15;
  int nc = blockIdx.x >> 4;
  int ko = (lane < 16) ? 0 : 8;
#pragma unroll
  for (int q = 0; q < 4; ++q) {
    int kr = wave * 4 + q;
    int i  = kc * 32 + kr;                      // Whh row (i)
    size_t flat = (size_t)i * kH3 + nc * 16 + jl;   // flat psi column for this lane
    v16bf bt;
#pragma unroll
    for (int u = 0; u < 8; ++u) {
      int k = ko + u;
      bt[u] = (k < kNLR) ? (bf16)W3[(size_t)k * kNPSI + flat] : (bf16)0.0f;
    }
#pragma unroll
    for (int u = 0; u < 8; ++u) {
      int k = ko + 16 + u;
      bt[8 + u] = (k < kNLR) ? (bf16)W3[(size_t)k * kNPSI + flat] : (bf16)0.0f;
    }
    float bias = b3[flat];
    int h     = (kr & 7) | ((kr & 16) >> 1);
    int lane2 = jl + ((kr & 8) ? 16 : 0);
    int stoff = lane2 * 16 + h;
#pragma unroll
    for (int mt = 0; mt < 4; ++mt) {
      v16bf at = load_a16(r_bf, mt * 16, 0, 32);
      v8f c = {};
      c = wmma_bf(at, bt, c);
#pragma unroll
      for (int v = 0; v < 8; ++v) {
        int bb = mt * 16 + v + ((lane < 16) ? 0 : 8);
        stage[bb * kStagePitch + stoff] = (bf16)(c[v] + bias);
      }
    }
  }
  __syncthreads();
  // coalesced writeout: 64 samples x 512 halves (= 64 x 64 16B-chunks)
  size_t stBase = (size_t)(nc * 16 + kc) * 512;
  const uint4* src = (const uint4*)stage;       // kStagePitch/8 = 65 chunks per sample row
#pragma unroll
  for (int it = 0; it < 16; ++it) {
    int linear = it * 256 + threadIdx.x;        // 4096 chunks
    int bb  = linear >> 6;
    int rem = linear & 63;
    uint4 v = src[bb * (kStagePitch / 8) + rem];
    *(uint4*)(whhswz + (size_t)bb * kWhhElems + stBase + rem * 8) = v;
  }
}

// ---- kernel 2b: psi tail (bh, Wih, C, d) -> compact psi2, linear stores ----
// block=128 (4 waves), grid=2073 ; tiles 49152..57443
__global__ void psi_rest_kernel(const bf16* r_bf, const float* W3, const float* b3,
                                bf16* psi2) {
  int wave = threadIdx.x >> 5;
  int lane = threadIdx.x & 31;
  int tile = kWhhTiles + blockIdx.x * 4 + wave;
  int n0 = tile * 16;
  int nl = lane & 15;
  int nn = n0 + nl;
  int ko = (lane < 16) ? 0 : 8;
  v16bf bt;
#pragma unroll
  for (int i = 0; i < 8; ++i) {
    int k = ko + i;
    bt[i] = (k < kNLR) ? (bf16)W3[(size_t)k * kNPSI + nn] : (bf16)0.0f;
  }
#pragma unroll
  for (int i = 0; i < 8; ++i) {
    int k = ko + 16 + i;
    bt[8 + i] = (k < kNLR) ? (bf16)W3[(size_t)k * kNPSI + nn] : (bf16)0.0f;
  }
  float bias = b3[nn];
#pragma unroll
  for (int mt = 0; mt < 4; ++mt) {
    v16bf at = load_a16(r_bf, mt * 16, 0, 32);
    v8f c = {};
    c = wmma_bf(at, bt, c);
#pragma unroll
    for (int v = 0; v < 8; ++v) {
      int bb = mt * 16 + v + ((lane < 16) ? 0 : 8);
      psi2[(size_t)bb * kPsi2 + (nn - OFF_BH)] = (bf16)(c[v] + bias);
    }
  }
}

// ---- kernel 3: gx[b,t,j] = inputs @ Wih (+bias) ----
// block=256 (8 waves), grid=6144
__global__ void gx_kernel(const bf16* inbf, const bf16* wbase, long wStride,
                          int sK, int sN, const bf16* biasBf, long biasStride,
                          const float* biasF, bf16* gx) {
  int wave = threadIdx.x >> 5;
  int lane = threadIdx.x & 31;
  int id = blockIdx.x * 8 + wave;       // 64*8*96 = 49152
  int b = id / (8 * 96);
  int rem = id % (8 * 96);
  int mt = rem / 96, nt = rem % 96;
  const bf16* A  = inbf + (size_t)b * kT * kIN;
  const bf16* Wp = wbase + (size_t)b * wStride;
  v8f c = {};
#pragma unroll
  for (int kc = 0; kc < 2; ++kc) {
    v16bf a  = load_a16(A, mt * 16, kc * 32, kIN);
    v16bf bt = load_b16(Wp, kc * 32, nt * 16, sK, sN);
    c = wmma_bf(a, bt, c);
  }
  int n = nt * 16 + (lane & 15);
  float bias = biasBf ? (float)biasBf[(size_t)b * biasStride + n] : biasF[n];
  bf16* outp = gx + (size_t)b * kT * kH3;
#pragma unroll
  for (int v = 0; v < 8; ++v) {
    int t = mt * 16 + v + ((lane < 16) ? 0 : 8);
    outp[(size_t)t * kH3 + n] = (bf16)(c[v] + bias);
  }
}

// ---- kernel 4: GRU recurrence (GEMV via WMMA, swizzled weights, 32 waves/sample) ----
__global__ void __launch_bounds__(1024) rec_kernel(
    const bf16* gx, const bf16* wbase, long wStride,
    const float* ghBias, const float* h0, bf16* hs) {
  __shared__ float ghs[kH3];
  __shared__ float hf[kH];
  __shared__ bf16  hb[kH];
  int b = blockIdx.x;
  int tid = threadIdx.x;
  int wave = tid >> 5, lane = tid & 31;
  if (tid < kH) {
    float v = h0 ? h0[(size_t)b * kH + tid] : 0.0f;
    hf[tid] = v; hb[tid] = (bf16)v;
  }
  __syncthreads();
  const bf16* Wp  = wbase + (size_t)b * wStride;
  const bf16* gxb = gx + (size_t)b * kT * kH3;
  bf16* hsb = hs + (size_t)b * kT * kH;
  int nc0 = wave * 3;                   // 3 x 16-column chunks per wave
  for (int t = 0; t < kT; ++t) {
    v8f acc0 = {}, acc1 = {}, acc2 = {};
#pragma unroll 4
    for (int kc = 0; kc < 16; ++kc) {
      int kb = kc * 32;
      v16bf a;
#pragma unroll
      for (int i = 0; i < 16; ++i) a[i] = (bf16)0.0f;
      if (lane == 0) {
#pragma unroll
        for (int i = 0; i < 8; ++i) { a[i] = hb[kb + i];     a[8 + i] = hb[kb + 16 + i]; }
      } else if (lane == 16) {
#pragma unroll
        for (int i = 0; i < 8; ++i) { a[i] = hb[kb + 8 + i]; a[8 + i] = hb[kb + 24 + i]; }
      }
      v16bf b0 = load_b_swz(Wp, kc, nc0);      acc0 = wmma_bf(a, b0, acc0);
      v16bf b1 = load_b_swz(Wp, kc, nc0 + 1);  acc1 = wmma_bf(a, b1, acc1);
      v16bf b2 = load_b_swz(Wp, kc, nc0 + 2);  acc2 = wmma_bf(a, b2, acc2);
    }
    if (lane < 16) {                     // D row 0 = c[0] of lanes 0..15
      int j = nc0 * 16 + lane;
      ghs[j]      = acc0[0] + (ghBias ? ghBias[j]      : 0.0f);
      ghs[j + 16] = acc1[0] + (ghBias ? ghBias[j + 16] : 0.0f);
      ghs[j + 32] = acc2[0] + (ghBias ? ghBias[j + 32] : 0.0f);
    }
    __syncthreads();
    if (tid < kH) {
      int i = tid;
      const bf16* g = gxb + (size_t)t * kH3;
      float r   = sigf((float)g[i] + ghs[i]);
      float z   = sigf((float)g[kH + i] + ghs[kH + i]);
      float eta = tanhf((float)g[2 * kH + i] + r * ghs[2 * kH + i]);
      float hn  = z * hf[i] + (1.0f - z) * eta;
      hf[i] = hn; hb[i] = (bf16)hn;
      hsb[(size_t)t * kH + i] = (bf16)hn;
    }
    __syncthreads();
  }
}

// ---- kernel 5: out = hs1@C1[b] + d1 + hs2@C2 + d2 + 2*inputs ----
__global__ void out_kernel(const bf16* hs1, const bf16* psi2, const bf16* hs2,
                           const bf16* c2bf, const float* d2, const float* inputs,
                           float* out) {
  int wave = threadIdx.x >> 5, lane = threadIdx.x & 31;
  int id = blockIdx.x * 8 + wave;       // 64*8*4 = 2048
  int b = id / 32;
  int rem = id % 32;
  int mt = rem / 4, nt = rem % 4;
  const bf16* A1 = hs1 + (size_t)b * kT * kH;
  const bf16* A2 = hs2 + (size_t)b * kT * kH;
  const bf16* C1 = psi2 + (size_t)b * kPsi2 + P2_C;
  v8f c = {};
#pragma unroll 4
  for (int kc = 0; kc < 16; ++kc) {
    v16bf a1 = load_a16(A1, mt * 16, kc * 32, kH);
    v16bf b1 = load_b16(C1, kc * 32, nt * 16, kOUT, 1);
    c = wmma_bf(a1, b1, c);
    v16bf a2 = load_a16(A2, mt * 16, kc * 32, kH);
    v16bf b2 = load_b16(c2bf, kc * 32, nt * 16, kOUT, 1);
    c = wmma_bf(a2, b2, c);
  }
  int o = nt * 16 + (lane & 15);
  float dd = (float)psi2[(size_t)b * kPsi2 + P2_DV + o] + d2[o];
#pragma unroll
  for (int v = 0; v < 8; ++v) {
    int t = mt * 16 + v + ((lane < 16) ? 0 : 8);
    size_t idx = ((size_t)b * kT + t) * kOUT + o;
    out[idx] = c[v] + dd + 2.0f * inputs[idx];
  }
}

extern "C" void kernel_launch(void* const* d_in, const int* in_sizes, int n_in,
                              void* d_out, int out_size, void* d_ws, size_t ws_size,
                              hipStream_t stream) {
  const float* inputs = (const float*)d_in[0];
  const float* mu     = (const float*)d_in[1];
  const float* sd     = (const float*)d_in[2];
  const float* eps    = (const float*)d_in[3];
  const float* state  = (const float*)d_in[4];
  const float* W1     = (const float*)d_in[5];
  const float* b1     = (const float*)d_in[6];
  const float* W2     = (const float*)d_in[7];
  const float* b2     = (const float*)d_in[8];
  const float* W3     = (const float*)d_in[9];
  const float* b3     = (const float*)d_in[10];
  const float* Wih2   = (const float*)d_in[11];
  const float* Whh2   = (const float*)d_in[12];
  const float* bih2   = (const float*)d_in[13];
  const float* bhh2   = (const float*)d_in[14];
  const float* C2     = (const float*)d_in[15];
  const float* d2v    = (const float*)d_in[16];
  float* out = (float*)d_out;
  (void)in_sizes; (void)n_in; (void)out_size; (void)ws_size;

  char* ws = (char*)d_ws;
  size_t off = 0;
  auto take = [&](size_t bytes) -> char* {
    char* p = ws + off;
    off = (off + bytes + 255) & ~(size_t)255;
    return p;
  };
  bf16* r_bf    = (bf16*)take((size_t)kB * 32 * sizeof(bf16));
  bf16* whh1swz = (bf16*)take((size_t)kB * kWhhElems * sizeof(bf16)); // 96 MB
  bf16* whh2swz = (bf16*)take((size_t)kWhhElems * sizeof(bf16));      // 1.5 MB
  bf16* psi2    = (bf16*)take((size_t)kB * kPsi2 * sizeof(bf16));     // 17 MB
  bf16* in_bf   = (bf16*)take((size_t)kB * kT * kIN * sizeof(bf16));
  bf16* wih2bf  = (bf16*)take((size_t)kH3 * kIN * sizeof(bf16));
  bf16* c2bf    = (bf16*)take((size_t)kH * kOUT * sizeof(bf16));
  bf16* gx1     = (bf16*)take((size_t)kB * kT * kH3 * sizeof(bf16));  // 25 MB
  bf16* gx2     = (bf16*)take((size_t)kB * kT * kH3 * sizeof(bf16));  // 25 MB
  bf16* hs1     = (bf16*)take((size_t)kB * kT * kH * sizeof(bf16));   // 8.4 MB
  bf16* hs2     = (bf16*)take((size_t)kB * kT * kH * sizeof(bf16));   // 8.4 MB

  hyper_kernel<<<dim3(kB), dim3(64), 0, stream>>>(mu, sd, eps, W1, b1, W2, b2, r_bf);
  cvt_kernel<<<dim3(512), dim3(256), 0, stream>>>(inputs, in_bf, kB * kT * kIN);
  cvt_kernel<<<dim3(128), dim3(256), 0, stream>>>(Wih2, wih2bf, kH3 * kIN);
  cvt_kernel<<<dim3(64),  dim3(256), 0, stream>>>(C2, c2bf, kH * kOUT);
  repack2_kernel<<<dim3(768), dim3(256), 0, stream>>>(Whh2, whh2swz);

  psi_whh_kernel<<<dim3(1536), dim3(256), 0, stream>>>(r_bf, W3, b3, whh1swz);
  psi_rest_kernel<<<dim3(2073), dim3(128), 0, stream>>>(r_bf, W3, b3, psi2);

  // gx1: per-sample Wih region of psi2 (sK=1536, sN=1), bias bh (bf16, per-sample)
  gx_kernel<<<dim3(6144), dim3(256), 0, stream>>>(
      in_bf, psi2 + P2_WIH, (long)kPsi2, kH3, 1,
      psi2 + P2_BH, (long)kPsi2, (const float*)nullptr, gx1);
  // gx2: shared Wih2^T (B[k=i][n=j] = Wih2[j][i] -> sK=1, sN=64), bias bih2 (fp32)
  gx_kernel<<<dim3(6144), dim3(256), 0, stream>>>(
      in_bf, wih2bf, 0L, 1, kIN,
      (const bf16*)nullptr, 0L, bih2, gx2);

  // recurrence 1: per-sample swizzled Whh, no gh bias, h0 = state
  rec_kernel<<<dim3(kB), dim3(1024), 0, stream>>>(
      gx1, whh1swz, (long)kWhhElems, (const float*)nullptr, state, hs1);
  // recurrence 2: shared swizzled Whh2, gh bias = bhh2 (inside reset gate), h0 = 0
  rec_kernel<<<dim3(kB), dim3(1024), 0, stream>>>(
      gx2, whh2swz, 0L, bhh2, (const float*)nullptr, hs2);

  out_kernel<<<dim3(256), dim3(256), 0, stream>>>(hs1, psi2, hs2, c2bf, d2v, inputs, out);
}
